// PCTDft_83056077570832
// MI455X (gfx1250) — matrix operand order repaired and checked by reading, not compiled, not run on hardware
//
#include <hip/hip_runtime.h>
#include <hip/hip_bf16.h>

// ---------------- problem constants ----------------
#define B_   8
#define N_   1024
#define K_   20
#define M_   (B_*N_)        // 8192 points
#define MK_  (M_*K_)        // 163840
#define S_   11
#define NSEG 50

typedef __attribute__((ext_vector_type(16))) _Float16 v16h;
typedef __attribute__((ext_vector_type(4)))  _Float16 h4;
typedef __attribute__((ext_vector_type(8)))  float    v8f;

// ---------------- WMMA GEMM: Y(Mo x Mn) = W(Mo x Kc) * X(Kc x Mn) ----------------
// f32 in global, f16 LDS tiles, f32 accumulation via v_wmma_f32_16x16x32_f16.
// Block tile 64x128x32, 8 waves (2x4), each wave computes 32x32 (2x2 frags ->
// 4 WMMAs per K-step). Double-buffered LDS: next tile's global loads are staged
// in registers during the WMMAs, one barrier per K-step. Optional per-row bias
// and fused activation (0 none, 1 lrelu 0.2, 2 relu). Strides allow batched views.
#define BMT 64
#define BNT 128
#define BKT 32

__global__ __launch_bounds__(256)
void k_gemm(float* __restrict__ Y, const float* __restrict__ W,
            const float* __restrict__ X, const float* __restrict__ bias,
            int Mo, int Kc, int Mn, long ldy, long ldw, long ldx, int act)
{
    __shared__ _Float16 sA[2][BMT][BKT];      // [buf][m][k]
    __shared__ _Float16 sB[2][BNT][BKT];      // [buf][n][k] (transposed)

    const int tid  = threadIdx.x;
    const int row0 = blockIdx.y * BMT;
    const int col0 = blockIdx.x * BNT;
    const int lane = tid & 31;
    const int wid  = tid >> 5;             // 0..7
    const int wr   = wid >> 2;             // 0..1  -> 32-row group
    const int wc   = wid & 3;              // 0..3  -> 32-col group
    const int nl   = lane & 15;
    const int hi   = lane >> 4;

    // fast-path eligibility (uniform): vectorizable global loads
    const bool alA = ((ldw & 3) == 0) && ((((unsigned long)W) & 15) == 0) &&
                     ((row0 + BMT) <= Mo);
    const bool alB = ((ldx & 3) == 0) && ((((unsigned long)X) & 15) == 0) &&
                     ((col0 + BNT) <= Mn);

    float stA[8];     // per-thread staging: A tile is 2048 elems / 256 thr
    float stB[16];    // B tile is 4096 elems / 256 thr

    // ---- load next tile from global into registers (no LDS traffic) ----
    auto loadA = [&](int k0, bool kFull) {
        if (alA && kFull) {
            #pragma unroll
            for (int r = 0; r < 2; ++r) {                    // 2 quads
                int q = tid + r * 256;
                int m = q >> 3, kq = (q & 7) * 4;
                const float4 v = *(const float4*)(W + (long)(row0 + m) * ldw + k0 + kq);
                stA[4*r+0] = v.x; stA[4*r+1] = v.y; stA[4*r+2] = v.z; stA[4*r+3] = v.w;
            }
        } else {
            #pragma unroll
            for (int j = 0; j < 8; ++j) {                    // 8 scalars, clamped
                int i = tid + j * 256;
                int m = i >> 5, k = i & 31;
                int gr = row0 + m, gk = k0 + k;
                bool ok = (gr < Mo) && (gk < Kc);
                long idx = (long)(ok ? gr : 0) * ldw + (ok ? gk : 0);
                float v = W[idx];                            // unconditional
                stA[j] = ok ? v : 0.0f;
            }
        }
    };
    auto loadB = [&](int k0, bool kFull) {
        if (alB && kFull) {
            #pragma unroll
            for (int r = 0; r < 4; ++r) {                    // 4 quads
                int q = tid + r * 256;
                int k = q >> 5, nq = (q & 31) * 4;
                const float4 v = *(const float4*)(X + (long)(k0 + k) * ldx + col0 + nq);
                stB[4*r+0] = v.x; stB[4*r+1] = v.y; stB[4*r+2] = v.z; stB[4*r+3] = v.w;
            }
        } else {
            #pragma unroll
            for (int j = 0; j < 16; ++j) {
                int i = tid + j * 256;
                int k = i >> 7, n = i & 127;
                int gk = k0 + k, gc = col0 + n;
                bool ok = (gk < Kc) && (gc < Mn);
                long idx = (long)(ok ? gk : 0) * ldx + (ok ? gc : 0);
                float v = X[idx];
                stB[j] = ok ? v : 0.0f;
            }
        }
    };
    // ---- convert staged registers to f16 and write into LDS buffer ----
    auto storeA = [&](int buf, bool kFull) {
        if (alA && kFull) {
            #pragma unroll
            for (int r = 0; r < 2; ++r) {
                int q = tid + r * 256;
                int m = q >> 3, kq = (q & 7) * 4;
                h4 h; h[0] = (_Float16)stA[4*r+0]; h[1] = (_Float16)stA[4*r+1];
                      h[2] = (_Float16)stA[4*r+2]; h[3] = (_Float16)stA[4*r+3];
                *(h4*)&sA[buf][m][kq] = h;
            }
        } else {
            #pragma unroll
            for (int j = 0; j < 8; ++j) {
                int i = tid + j * 256;
                int m = i >> 5, k = i & 31;
                sA[buf][m][k] = (_Float16)stA[j];
            }
        }
    };
    auto storeB = [&](int buf, bool kFull) {
        if (alB && kFull) {
            #pragma unroll
            for (int r = 0; r < 4; ++r) {
                int q = tid + r * 256;
                int k = q >> 5, nq = (q & 31) * 4;
                sB[buf][nq + 0][k] = (_Float16)stB[4*r+0];
                sB[buf][nq + 1][k] = (_Float16)stB[4*r+1];
                sB[buf][nq + 2][k] = (_Float16)stB[4*r+2];
                sB[buf][nq + 3][k] = (_Float16)stB[4*r+3];
            }
        } else {
            #pragma unroll
            for (int j = 0; j < 16; ++j) {
                int i = tid + j * 256;
                int k = i >> 7, n = i & 127;
                sB[buf][n][k] = (_Float16)stB[j];
            }
        }
    };

    v8f acc00 = {}, acc01 = {}, acc10 = {}, acc11 = {};

    const int nT = (Kc + BKT - 1) / BKT;
    // prologue: tile 0 -> buffer 0
    {
        bool kFull = BKT <= Kc;
        loadA(0, kFull);  loadB(0, kFull);
        storeA(0, kFull); storeB(0, kFull);
    }
    __syncthreads();

    for (int t = 0; t < nT; ++t) {
        const int  cur   = t & 1;
        const bool more  = (t + 1) < nT;
        const bool kFull = ((t + 1) * BKT + BKT) <= Kc;   // next tile fully in K
        if (more) { loadA((t + 1) * BKT, kFull); loadB((t + 1) * BKT, kFull); }

        // ---- fragments (ISA 7.12.2 layouts) from buffer `cur` ----
        v16h a0, a1, b0, b1;
        #pragma unroll
        for (int e = 0; e < 16; ++e) {
            // A: lane row m=nl; halves cover k = hi*8+e (e<8), 16+hi*8+(e-8) (e>=8)
            int ka = (e < 8) ? (hi * 8 + e) : (16 + hi * 8 + (e - 8));
            a0[e] = sA[cur][wr * 32 + nl][ka];
            a1[e] = sA[cur][wr * 32 + 16 + nl][ka];
            // B: lane col n=nl; same k order as A
            int kb = ka;
            b0[e] = sB[cur][wc * 32 + nl][kb];
            b1[e] = sB[cur][wc * 32 + 16 + nl][kb];
        }
        acc00 = __builtin_amdgcn_wmma_f32_16x16x32_f16(false, a0, false, b0, (short)0, acc00, false, false);
        acc01 = __builtin_amdgcn_wmma_f32_16x16x32_f16(false, a0, false, b1, (short)0, acc01, false, false);
        acc10 = __builtin_amdgcn_wmma_f32_16x16x32_f16(false, a1, false, b0, (short)0, acc10, false, false);
        acc11 = __builtin_amdgcn_wmma_f32_16x16x32_f16(false, a1, false, b1, (short)0, acc11, false, false);

        if (more) { storeA(cur ^ 1, kFull); storeB(cur ^ 1, kFull); }
        __syncthreads();
    }

    // ---- epilogue: D layout lane L: n = L&15 ; m = vgpr + (L>=16 ? 8 : 0) ----
    #pragma unroll
    for (int fi = 0; fi < 2; ++fi) {
        #pragma unroll
        for (int fj = 0; fj < 2; ++fj) {
            const v8f* accp = fi == 0 ? (fj == 0 ? &acc00 : &acc01)
                                      : (fj == 0 ? &acc10 : &acc11);
            int coln = col0 + wc * 32 + fj * 16 + nl;
            #pragma unroll
            for (int v = 0; v < 8; ++v) {
                int rowm = row0 + wr * 32 + fi * 16 + v + hi * 8;
                if (rowm < Mo && coln < Mn) {
                    float o = (*accp)[v];
                    if (bias) o += bias[rowm];
                    if (act == 1)      o = (o >= 0.f) ? o : 0.2f * o;
                    else if (act == 2) o = fmaxf(o, 0.f);
                    Y[(long)rowm * ldy + coln] = o;
                }
            }
        }
    }
}

// ---------------- training-mode BatchNorm (per-channel-row stats) ----------------
__global__ __launch_bounds__(256)
void k_bn_stats(const float* __restrict__ Y, float* __restrict__ mean,
                float* __restrict__ var, long ld, long Mn)
{
    __shared__ float ss[256], sq[256];
    long r = blockIdx.x;
    const float* p = Y + r * ld;
    float s = 0.f, q = 0.f;
    for (long i = threadIdx.x; i < Mn; i += 256) { float v = p[i]; s += v; q += v * v; }
    ss[threadIdx.x] = s; sq[threadIdx.x] = q;
    __syncthreads();
    for (int o = 128; o > 0; o >>= 1) {
        if ((int)threadIdx.x < o) { ss[threadIdx.x] += ss[threadIdx.x + o];
                                    sq[threadIdx.x] += sq[threadIdx.x + o]; }
        __syncthreads();
    }
    if (threadIdx.x == 0) {
        float m = ss[0] / (float)Mn;
        mean[r] = m;
        var[r]  = fmaxf(sq[0] / (float)Mn - m * m, 0.f);
    }
}

__global__ void k_bn_apply(float* __restrict__ Y, const float* __restrict__ mean,
                           const float* __restrict__ var, const float* __restrict__ g,
                           const float* __restrict__ b, long ld, long Mn, int rows, int act)
{
    long idx = (long)blockIdx.x * blockDim.x + threadIdx.x;
    long total = (long)rows * Mn;
    if (idx >= total) return;
    long r = idx / Mn, c = idx % Mn;
    float y = (Y[r * ld + c] - mean[r]) * rsqrtf(var[r] + 1e-5f) * g[r] + b[r];
    if (act == 1)      y = (y >= 0.f) ? y : 0.2f * y;
    else if (act == 2) y = fmaxf(y, 0.f);
    Y[r * ld + c] = y;
}

// ---------------- kNN (K=20 nearest incl. self) -> feat (6, B*N*K) ----------------
__global__ __launch_bounds__(256)
void k_feat(const float* __restrict__ x, float* __restrict__ feat)
{
    __shared__ float d2s[N_];
    __shared__ float bval[256];
    __shared__ int   bidx[256];
    __shared__ int   sel[K_];
    int b = blockIdx.x / N_;
    int n = blockIdx.x % N_;
    const float* xb = x + (long)b * 3 * N_;
    float px = xb[n], py = xb[N_ + n], pz = xb[2 * N_ + n];
    for (int j = threadIdx.x; j < N_; j += 256) {
        float dx = xb[j] - px, dy = xb[N_ + j] - py, dz = xb[2 * N_ + j] - pz;
        d2s[j] = dx * dx + dy * dy + dz * dz;
    }
    __syncthreads();
    for (int t = 0; t < K_; ++t) {
        float bv = 1e30f; int bi = 0x7fffffff;
        for (int j = threadIdx.x; j < N_; j += 256) {
            float v = d2s[j];
            if (v < bv || (v == bv && j < bi)) { bv = v; bi = j; }
        }
        bval[threadIdx.x] = bv; bidx[threadIdx.x] = bi;
        __syncthreads();
        for (int o = 128; o > 0; o >>= 1) {
            if ((int)threadIdx.x < o) {
                float v2 = bval[threadIdx.x + o]; int i2 = bidx[threadIdx.x + o];
                if (v2 < bval[threadIdx.x] ||
                    (v2 == bval[threadIdx.x] && i2 < bidx[threadIdx.x])) {
                    bval[threadIdx.x] = v2; bidx[threadIdx.x] = i2;
                }
            }
            __syncthreads();
        }
        if (threadIdx.x == 0) { sel[t] = bidx[0]; d2s[bidx[0]] = 1e30f; }
        __syncthreads();
    }
    if (threadIdx.x < K_) {
        int k = threadIdx.x, j = sel[k];
        float nx = xb[j], ny = xb[N_ + j], nz = xb[2 * N_ + j];
        long col = ((long)(b * N_ + n)) * K_ + k;
        feat[0 * (long)MK_ + col] = nx - px;
        feat[1 * (long)MK_ + col] = ny - py;
        feat[2 * (long)MK_ + col] = nz - pz;
        feat[3 * (long)MK_ + col] = px;
        feat[4 * (long)MK_ + col] = py;
        feat[5 * (long)MK_ + col] = pz;
    }
}

// ---------------- small reductions / reshapes ----------------
__global__ void k_maxk(const float* __restrict__ in, float* __restrict__ out, int rows)
{
    long idx = (long)blockIdx.x * blockDim.x + threadIdx.x;
    long total = (long)rows * M_;
    if (idx >= total) return;
    long r = idx / M_, i = idx % M_;
    const float* p = in + r * (long)MK_ + i * K_;
    float m = p[0];
    for (int k = 1; k < K_; ++k) m = fmaxf(m, p[k]);
    out[r * (long)M_ + i] = m;
}

__global__ void k_maxn(const float* __restrict__ in, float* __restrict__ out, int rows)
{
    int idx = blockIdx.x * blockDim.x + threadIdx.x;   // r*B + b
    if (idx >= rows * B_) return;
    int r = idx / B_, b = idx % B_;
    const float* p = in + (long)r * M_ + b * N_;
    float m = p[0];
    for (int n = 1; n < N_; ++n) m = fmaxf(m, p[n]);
    out[idx] = m;
}

__global__ void k_transform(const float* __restrict__ x, const float* __restrict__ T9,
                            float* __restrict__ xt)
{
    int idx = blockIdx.x * blockDim.x + threadIdx.x;   // over 3*M
    if (idx >= 3 * M_) return;
    int d = idx / M_, m = idx % M_;
    int b = m / N_, n = m % N_;
    const float* xb = x + (long)b * 3 * N_;
    float s = 0.f;
    for (int c = 0; c < 3; ++c) s += xb[c * N_ + n] * T9[(c * 3 + d) * B_ + b];
    xt[(long)d * M_ + m] = s;
}

// ---------------- voxel occupancy grid (11^3 per point) ----------------
__global__ __launch_bounds__(256)
void k_grid(const float* __restrict__ xt, float* __restrict__ grid)
{
    __shared__ float g[S_ * S_ * S_];
    int p = blockIdx.x;                 // b*N+n
    for (int i = threadIdx.x; i < S_ * S_ * S_; i += 256) g[i] = 0.f;
    __syncthreads();
    int b = p / N_;
    float cx = xt[0 * (long)M_ + p], cy = xt[1 * (long)M_ + p], cz = xt[2 * (long)M_ + p];
    const float r     = 3.872983346207417f;        // sqrt(15)
    const float inv2r = 1.0f / (2.0f * r);
    for (int j = threadIdx.x; j < N_; j += 256) {
        long q = (long)b * N_ + j;
        float dx = xt[0 * (long)M_ + q] - cx;
        float dy = xt[1 * (long)M_ + q] - cy;
        float dz = xt[2 * (long)M_ + q] - cz;
        float d2 = dx * dx + dy * dy + dz * dz;
        if (d2 <= 15.0f) {
            int ix = (int)floorf((dx + r) * inv2r * (float)S_);
            int iy = (int)floorf((dy + r) * inv2r * (float)S_);
            int iz = (int)floorf((dz + r) * inv2r * (float)S_);
            ix = ix < 0 ? 0 : (ix > S_ - 1 ? S_ - 1 : ix);
            iy = iy < 0 ? 0 : (iy > S_ - 1 ? S_ - 1 : iy);
            iz = iz < 0 ? 0 : (iz > S_ - 1 ? S_ - 1 : iz);
            atomicAdd(&g[(ix * S_ + iy) * S_ + iz], 1.0f);
        }
    }
    __syncthreads();
    for (int i = threadIdx.x; i < S_ * S_ * S_; i += 256)
        grid[(long)p * (S_ * S_ * S_) + i] = g[i];
}

// ---------------- conv3d stage 1: 1 -> 8 @ 5^3, 11^3 -> 7^3 ----------------
__global__ __launch_bounds__(256)
void k_conv3d_s1(const float* __restrict__ grid, const float* __restrict__ w,
                 const float* __restrict__ bias, float* __restrict__ out)
{
    __shared__ float in[1331];
    __shared__ float wt[8 * 125];
    int p = blockIdx.x;
    for (int i = threadIdx.x; i < 1331; i += 256) in[i] = grid[(long)p * 1331 + i];
    for (int i = threadIdx.x; i < 1000; i += 256) wt[i] = w[i];
    __syncthreads();
    const long PM1 = (long)M_ * 343;
    for (int o = threadIdx.x; o < 8 * 343; o += 256) {
        int c = o / 343, s = o % 343;
        int ox = s / 49, oy = (s / 7) % 7, oz = s % 7;
        float acc = bias[c];
        const float* wc = wt + c * 125;
        for (int dx = 0; dx < 5; ++dx)
            for (int dy = 0; dy < 5; ++dy)
                for (int dz = 0; dz < 5; ++dz)
                    acc += in[(ox + dx) * 121 + (oy + dy) * 11 + (oz + dz)] *
                           wc[(dx * 5 + dy) * 5 + dz];
        out[(long)c * PM1 + (long)p * 343 + s] = acc;
    }
}

// ---------------- conv3d stage 2: 8 -> 64 @ 5^3, 7^3 -> 3^3 ----------------
__global__ __launch_bounds__(256)
void k_conv3d_s2(const float* __restrict__ s1, const float* __restrict__ w,
                 const float* __restrict__ bias, float* __restrict__ out)
{
    __shared__ float in[8 * 343];
    int p = blockIdx.x;
    const long PM1 = (long)M_ * 343;
    for (int i = threadIdx.x; i < 8 * 343; i += 256) {
        int c = i / 343, s = i % 343;
        in[i] = s1[(long)c * PM1 + (long)p * 343 + s];
    }
    __syncthreads();
    const long PM2 = (long)M_ * 27;
    for (int o = threadIdx.x; o < 64 * 27; o += 256) {
        int co = o / 27, s = o % 27;
        int ox = s / 9, oy = (s / 3) % 3, oz = s % 3;
        float acc = bias[co];
        for (int ci = 0; ci < 8; ++ci) {
            const float* wp = w + ((long)co * 8 + ci) * 125;
            const float* ip = in + ci * 343;
            for (int dx = 0; dx < 5; ++dx)
                for (int dy = 0; dy < 5; ++dy)
                    for (int dz = 0; dz < 5; ++dz)
                        acc += ip[(ox + dx) * 49 + (oy + dy) * 7 + (oz + dz)] *
                               wp[(dx * 5 + dy) * 5 + dz];
        }
        out[(long)co * PM2 + (long)p * 27 + s] = acc;
    }
}

// flatten stage-2 output (64, p*27+s) -> X3 (1728, p) for the stage-3 GEMM
__global__ void k_im2col3(const float* __restrict__ s2, float* __restrict__ X3)
{
    long idx = (long)blockIdx.x * blockDim.x + threadIdx.x;
    if (idx >= (long)1728 * M_) return;
    long k = idx / M_, p = idx % M_;
    long c = k / 27, s = k % 27;
    X3[idx] = s2[c * (long)M_ * 27 + p * 27 + s];
}

// ---------------- elementwise helpers ----------------
__global__ void k_sub(const float* a, const float* b, float* y, long n)
{ long i = (long)blockIdx.x * blockDim.x + threadIdx.x; if (i < n) y[i] = a[i] - b[i]; }
__global__ void k_add(const float* a, const float* b, float* y, long n)
{ long i = (long)blockIdx.x * blockDim.x + threadIdx.x; if (i < n) y[i] = a[i] + b[i]; }

// Q (32, M) -> QT (B, N, 32)
__global__ void k_qt(const float* __restrict__ Q, float* __restrict__ QT)
{
    int idx = blockIdx.x * blockDim.x + threadIdx.x;
    if (idx >= 32 * M_) return;
    int c = idx / M_, m = idx % M_;
    int b = m / N_, i = m % N_;
    QT[((long)b * N_ + i) * 32 + c] = Q[(long)c * M_ + m];
}

// row-wise softmax of E/(sqrt(128)) over N; E: (B, N, N)
__global__ __launch_bounds__(256)
void k_softmax(float* __restrict__ E)
{
    __shared__ float red[256];
    long rb = blockIdx.x;                  // 0..B*N-1
    long b = rb / N_, i = rb % N_;
    float* p = E + b * (long)N_ * N_ + i * N_;
    const float inv = 0.08838834764831845f;   // 1/sqrt(128)
    float mx = -1e30f;
    for (int j = threadIdx.x; j < N_; j += 256) mx = fmaxf(mx, p[j] * inv);
    red[threadIdx.x] = mx; __syncthreads();
    for (int o = 128; o > 0; o >>= 1) {
        if ((int)threadIdx.x < o) red[threadIdx.x] = fmaxf(red[threadIdx.x], red[threadIdx.x + o]);
        __syncthreads();
    }
    mx = red[0]; __syncthreads();
    float s = 0.f;
    for (int j = threadIdx.x; j < N_; j += 256) {
        float e = __expf(p[j] * inv - mx);
        p[j] = e; s += e;
    }
    red[threadIdx.x] = s; __syncthreads();
    for (int o = 128; o > 0; o >>= 1) {
        if ((int)threadIdx.x < o) red[threadIdx.x] += red[threadIdx.x + o];
        __syncthreads();
    }
    float rs = 1.f / red[0];
    for (int j = threadIdx.x; j < N_; j += 256) p[j] *= rs;
}

// build (1536, M): rows 0..1023 broadcast per-batch max, rows 1024..1535 = xs
__global__ void k_cat(const float* __restrict__ gmax, const float* __restrict__ xs,
                      float* __restrict__ cat)
{
    long idx = (long)blockIdx.x * blockDim.x + threadIdx.x;
    if (idx >= (long)1536 * M_) return;
    long r = idx / M_, m = idx % M_;
    if (r < 1024) { int b = (int)(m / N_); cat[idx] = gmax[r * B_ + b]; }
    else          cat[idx] = xs[(r - 1024) * (long)M_ + m];
}

// (50, M) channel-major -> (B, 50, N)
__global__ void k_out(const float* __restrict__ Y, float* __restrict__ out)
{
    long idx = (long)blockIdx.x * blockDim.x + threadIdx.x;
    if (idx >= (long)B_ * NSEG * N_) return;
    long b = idx / ((long)NSEG * N_);
    long c = (idx / N_) % NSEG;
    long n = idx % N_;
    out[idx] = Y[c * (long)M_ + b * N_ + n];
}

// =====================================================================
extern "C" void kernel_launch(void* const* d_in, const int* in_sizes, int n_in,
                              void* d_out, int out_size, void* d_ws, size_t ws_size,
                              hipStream_t stream)
{
    (void)in_sizes; (void)n_in; (void)out_size; (void)ws_size;

    // ---- unpack inputs (setup_inputs() dict insertion order) ----
    int ii = 0;
    auto nxt = [&]() { return (const float*)d_in[ii++]; };
    const float* x   = nxt();
    const float *tw1 = nxt(), *tg1 = nxt(), *tb1 = nxt();
    const float *tw2 = nxt(), *tg2 = nxt(), *tb2 = nxt();
    const float *tw3 = nxt(), *tg3 = nxt(), *tb3 = nxt();
    const float *tl1 = nxt(), *tg4 = nxt(), *tb4 = nxt();
    const float *tl2 = nxt(), *tg5 = nxt(), *tb5 = nxt();
    const float *tl3 = nxt(), *tl3b = nxt();
    const float *w31 = nxt(), *b31 = nxt(), *g1 = nxt(), *bb1 = nxt();
    const float *w32 = nxt(), *b32 = nxt(), *g2 = nxt(), *bb2 = nxt();
    const float *w33 = nxt(), *b33 = nxt(), *g3 = nxt(), *bb3 = nxt();
    const float *w4  = nxt(), *b4  = nxt(), *g4 = nxt(), *bb4 = nxt();
    const float *w5  = nxt(), *b5  = nxt(), *g5 = nxt(), *bb5 = nxt();
    const float *w6  = nxt(), *b6  = nxt(), *g6 = nxt(), *bb6 = nxt();
    const float *w7  = nxt(), *b7  = nxt(), *g7 = nxt(), *bb7 = nxt();
    struct SAp { const float *wq,*wk,*wv,*bv,*wt,*bt,*g_an,*b_an,*wf1,*bf1,*wf2,*bf2,*g_ff,*b_ff; };
    SAp sa[4];
    for (int L = 0; L < 4; ++L) {
        sa[L].wq = nxt(); sa[L].wk = nxt(); sa[L].wv = nxt(); sa[L].bv = nxt();
        sa[L].wt = nxt(); sa[L].bt = nxt(); sa[L].g_an = nxt(); sa[L].b_an = nxt();
        sa[L].wf1 = nxt(); sa[L].bf1 = nxt(); sa[L].wf2 = nxt(); sa[L].bf2 = nxt();
        sa[L].g_ff = nxt(); sa[L].b_ff = nxt();
    }
    const float *wxx = nxt(), *gxx = nxt(), *bxx = nxt();       // w__, g__, b__
    const float *w5h = nxt(), *b5h = nxt(), *g5h = nxt(), *bb5h = nxt();
    const float *w6h = nxt(), *b6h = nxt(), *g6h = nxt(), *bb6h = nxt();
    const float *w7h = nxt(), *b7h = nxt();

    // ---- workspace carve-up (floats) ----
    float* wsp = (float*)d_ws;
    size_t off = 0;
    auto alloc = [&](size_t nf) { float* p = wsp + off; off += nf; return p; };
    const long PM1 = (long)M_ * 343;
    const long PM2 = (long)M_ * 27;

    float* A  = alloc(22478848); // arena A: tnh2 (128xMK) / conv3d stage1 (8xPM1)
    float* Bb = alloc(14155776); // arena B: tnh1 (64xMK) / X3 (1728xM) / cat (1536xM)
    float* C  = alloc(14155776); // arena C: tnh3 / grid / stage2 / attn E / head H1

    float* feat  = alloc((size_t)6 * MK_);
    float* h2max = alloc((size_t)128 * M_);
    float* P     = alloc(1024 * B_);
    float* F1    = alloc(512 * B_);
    float* F2    = alloc(256 * B_);
    float* T9    = alloc(128);
    float* xt    = alloc((size_t)3 * M_);
    float* G8    = alloc((size_t)8 * M_);
    float* emb   = alloc((size_t)160 * M_);
    float* h6    = alloc((size_t)128 * M_);
    float* x0    = alloc((size_t)128 * M_);
    float* xs    = alloc((size_t)512 * M_);
    float* Qb    = alloc((size_t)32 * M_);
    float* Kb    = alloc((size_t)32 * M_);
    float* Vb    = alloc((size_t)128 * M_);
    float* QT    = alloc((size_t)B_ * N_ * 32);
    float* xrb   = alloc((size_t)128 * M_);
    float* tA    = alloc((size_t)128 * M_);
    float* tB    = alloc((size_t)128 * M_);
    float* h512b = alloc((size_t)512 * M_);
    float* gmaxb = alloc(1024 * B_);
    float* H2    = alloc((size_t)512 * M_);
    float* H3    = alloc((size_t)256 * M_);
    float* Y50   = alloc((size_t)NSEG * M_);
    float* meanb = alloc(2048);
    float* varb  = alloc(2048);

    // ---- launch helpers ----
    auto gemm = [&](float* Y, const float* W, const float* X, const float* bias,
                    int Mo, int Kc, int Mn, long ldy, long ldw, long ldx, int act) {
        dim3 g((Mn + BNT - 1) / BNT, (Mo + BMT - 1) / BMT);
        k_gemm<<<g, dim3(256), 0, stream>>>(Y, W, X, bias, Mo, Kc, Mn, ldy, ldw, ldx, act);
    };
    auto bn = [&](float* Y, const float* g, const float* b, int rows, long ld, long Mn, int act) {
        k_bn_stats<<<dim3(rows), dim3(256), 0, stream>>>(Y, meanb, varb, ld, Mn);
        long total = (long)rows * Mn;
        k_bn_apply<<<dim3((unsigned)((total + 255) / 256)), dim3(256), 0, stream>>>(
            Y, meanb, varb, g, b, ld, Mn, rows, act);
    };
    auto ew = [&](long n) { return dim3((unsigned)((n + 255) / 256)); };

    // =========== transform_net ===========
    k_feat<<<B_ * N_, 256, 0, stream>>>(x, feat);
    float* tnh1 = Bb;
    gemm(tnh1, tw1, feat, nullptr, 64, 6, MK_, MK_, 6, MK_, 0);
    bn(tnh1, tg1, tb1, 64, MK_, MK_, 1);
    float* tnh2 = A;
    gemm(tnh2, tw2, tnh1, nullptr, 128, 64, MK_, MK_, 64, MK_, 0);
    bn(tnh2, tg2, tb2, 128, MK_, MK_, 1);
    k_maxk<<<ew((long)128 * M_), 256, 0, stream>>>(tnh2, h2max, 128);
    float* tnh3 = C;
    gemm(tnh3, tw3, h2max, nullptr, 1024, 128, M_, M_, 128, M_, 0);
    bn(tnh3, tg3, tb3, 1024, M_, M_, 1);
    k_maxn<<<ew(1024 * B_), 256, 0, stream>>>(tnh3, P, 1024);
    gemm(F1, tl1, P, nullptr, 512, 1024, B_, B_, 1024, B_, 0);
    bn(F1, tg4, tb4, 512, B_, B_, 1);
    gemm(F2, tl2, F1, nullptr, 256, 512, B_, B_, 512, B_, 0);
    bn(F2, tg5, tb5, 256, B_, B_, 1);
    gemm(T9, tl3, F2, tl3b, 9, 256, B_, B_, 256, B_, 0);
    k_transform<<<ew(3 * M_), 256, 0, stream>>>(x, T9, xt);

    // =========== input embedding ===========
    float* gridb = C;    // tnh3 dead
    k_grid<<<M_, 256, 0, stream>>>(xt, gridb);
    float* s1b = A;      // tnh2 dead
    k_conv3d_s1<<<M_, 256, 0, stream>>>(gridb, w31, b31, s1b);
    bn(s1b, g1, bb1, 8, PM1, PM1, 1);
    float* s2b = C;      // grid dead
    k_conv3d_s2<<<M_, 256, 0, stream>>>(s1b, w32, b32, s2b);
    bn(s2b, g2, bb2, 64, PM2, PM2, 1);
    float* X3 = Bb;      // tnh1 dead
    k_im2col3<<<ew((long)1728 * M_), 256, 0, stream>>>(s2b, X3);
    gemm(emb, w33, X3, b33, 128, 1728, M_, M_, 1728, M_, 0);   // local -> rows 0..127
    bn(emb, g3, bb3, 128, M_, M_, 1);
    gemm(G8, w4, xt, b4, 8, 3, M_, M_, 3, M_, 0);
    bn(G8, g4, bb4, 8, M_, M_, 1);
    gemm(emb + (long)128 * M_, w5, G8, b5, 32, 8, M_, M_, 8, M_, 0);  // glob -> rows 128..159
    bn(emb + (long)128 * M_, g5, bb5, 32, M_, M_, 1);
    gemm(h6, w6, emb, b6, 128, 160, M_, M_, 160, M_, 0);
    bn(h6, g6, bb6, 128, M_, M_, 1);
    gemm(x0, w7, h6, b7, 128, 128, M_, M_, 128, M_, 0);
    bn(x0, g7, bb7, 128, M_, M_, 1);

    // =========== 4 x offset self-attention layers ===========
    float* E = C;        // s2b dead after im2col
    auto sa_layer = [&](const float* xin, float* xout, const SAp& p) {
        gemm(Qb, p.wq, xin, nullptr, 32, 128, M_, M_, 128, M_, 0);
        gemm(Kb, p.wk, xin, nullptr, 32, 128, M_, M_, 128, M_, 0);
        gemm(Vb, p.wv, xin, p.bv, 128, 128, M_, M_, 128, M_, 0);
        k_qt<<<ew(32 * M_), 256, 0, stream>>>(Qb, QT);
        for (int b = 0; b < B_; ++b)
            gemm(E + (long)b * N_ * N_, QT + (long)b * N_ * 32, Kb + (long)b * N_,
                 nullptr, N_, 32, N_, N_, 32, M_, 0);
        k_softmax<<<B_ * N_, 256, 0, stream>>>(E);
        for (int b = 0; b < B_; ++b)
            gemm(xrb + (long)b * N_, Vb + (long)b * N_, E + (long)b * N_ * N_,
                 nullptr, 128, N_, N_, M_, M_, N_, 0);
        k_sub<<<ew((long)128 * M_), 256, 0, stream>>>(xin, xrb, tA, (long)128 * M_);
        gemm(tB, p.wt, tA, p.bt, 128, 128, M_, M_, 128, M_, 0);
        bn(tB, p.g_an, p.b_an, 128, M_, M_, 2);                 // relu
        k_add<<<ew((long)128 * M_), 256, 0, stream>>>(xin, tB, tA, (long)128 * M_); // xc
        gemm(h512b, p.wf1, tA, p.bf1, 512, 128, M_, M_, 128, M_, 2);  // fused relu
        gemm(tB, p.wf2, h512b, p.bf2, 128, 512, M_, M_, 512, M_, 0);
        k_add<<<ew((long)128 * M_), 256, 0, stream>>>(tA, tB, xout, (long)128 * M_);
        bn(xout, p.g_ff, p.b_ff, 128, M_, M_, 0);
    };
    sa_layer(x0,                    xs,                     sa[0]);
    sa_layer(xs,                    xs + (long)128 * M_,    sa[1]);
    sa_layer(xs + (long)128 * M_,   xs + (long)256 * M_,    sa[2]);
    sa_layer(xs + (long)256 * M_,   xs + (long)384 * M_,    sa[3]);

    // =========== segmentation head ===========
    float* H1 = C;       // E dead
    gemm(H1, wxx, xs, nullptr, 1024, 512, M_, M_, 512, M_, 0);
    bn(H1, gxx, bxx, 1024, M_, M_, 1);
    k_maxn<<<ew(1024 * B_), 256, 0, stream>>>(H1, gmaxb, 1024);
    float* catb = Bb;    // X3 dead
    k_cat<<<ew((long)1536 * M_), 256, 0, stream>>>(gmaxb, xs, catb);
    gemm(H2, w5h, catb, b5h, 512, 1536, M_, M_, 1536, M_, 0);
    bn(H2, g5h, bb5h, 512, M_, M_, 2);
    gemm(H3, w6h, H2, b6h, 256, 512, M_, M_, 512, M_, 0);
    bn(H3, g6h, bb6h, 256, M_, M_, 2);
    gemm(Y50, w7h, H3, b7h, NSEG, 256, M_, M_, 256, M_, 0);
    k_out<<<ew((long)B_ * NSEG * N_), 256, 0, stream>>>(Y50, (float*)d_out);
}
